// IntraModalityEnhance_30073361007269
// MI455X (gfx1250) — matrix operand order repaired and checked by reading, not compiled, not run on hardware
//
#include <hip/hip_runtime.h>
#include <math.h>

// Problem constants (from reference): B=2, S=2048, D=1024, H=16, HD=64, W=31, E=1024
#define B_ 2
#define S_ 2048
#define D_ 1024
#define H_ 16
#define HD_ 64
#define E_ 1024

typedef __bf16 bf16t;
typedef __attribute__((ext_vector_type(16))) __bf16 v16bf;
typedef __attribute__((ext_vector_type(8)))  __bf16 v8bf;
typedef __attribute__((ext_vector_type(8)))  float  v8f;
typedef __attribute__((ext_vector_type(4)))  float  v4f;
typedef __attribute__((ext_vector_type(4)))  unsigned int v4u;
typedef __attribute__((ext_vector_type(8)))  int v8i;
typedef __attribute__((ext_vector_type(4)))  int v4i;

#if __has_builtin(__builtin_amdgcn_tensor_load_to_lds)
#define HAVE_TDM 1
#else
#define HAVE_TDM 0
#endif

// ---------------------------------------------------------------------------
// Elementwise f32 -> bf16 (n multiple of 2048)
// ---------------------------------------------------------------------------
__global__ __launch_bounds__(256)
void cvt_f32_bf16(const float* __restrict__ src, bf16t* __restrict__ dst, int n)
{
    const int i = (blockIdx.x * 256 + threadIdx.x) * 8;
    if (i >= n) return;
    v4f a = *(const v4f*)(src + i);
    v4f b = *(const v4f*)(src + i + 4);
    v8bf o;
    o[0] = (bf16t)a.x; o[1] = (bf16t)a.y; o[2] = (bf16t)a.z; o[3] = (bf16t)a.w;
    o[4] = (bf16t)b.x; o[5] = (bf16t)b.y; o[6] = (bf16t)b.z; o[7] = (bf16t)b.w;
    *(v8bf*)(dst + i) = o;
}

// ---------------------------------------------------------------------------
// Per-head transpose + convert: src[h][k][n] (f32) -> dst[h][n][k] (bf16)
// blockIdx.y = head; K*N elements per head.
// ---------------------------------------------------------------------------
__global__ __launch_bounds__(256)
void cvt_transpose(const float* __restrict__ src, bf16t* __restrict__ dst, int K, int N)
{
    const long base = (long)blockIdx.y * K * N;
    const int gid = blockIdx.x * 256 + threadIdx.x;
    if (gid >= K * N) return;
    const int k = gid / N;
    const int n = gid - k * N;
    dst[base + (long)n * K + k] = (bf16t)src[base + gid];
}

// ---------------------------------------------------------------------------
// TDM: issue a 2-D tile load (tile_k x tile_rows, bf16) into LDS.
// LDS padding done by TDM: +16B after every 16 DWORDs (one 64B row) => 80B pitch,
// matching the [rows][40] bf16 LDS layout used by the WMMA fragments.
// ---------------------------------------------------------------------------
#if HAVE_TDM
__device__ __forceinline__ void tdm_load_2d(unsigned lds_off, const bf16t* gptr,
                                            unsigned tile_k, unsigned tile_rows,
                                            unsigned ld /* row stride, elements */)
{
    const unsigned long long ga = (unsigned long long)(size_t)gptr;
    v4u g0;
    g0[0] = 1u;                                            // count=1, user descriptor
    g0[1] = lds_off;                                       // lds_addr (bytes)
    g0[2] = (unsigned)(ga & 0xffffffffull);                // global_addr[31:0]
    g0[3] = (unsigned)((ga >> 32) & 0x1ffffffull)          // global_addr[56:32]
          | (2u << 30);                                    // type = 2 ("image")
    v8i g1;
    g1[0] = (int)((1u << 16)     // data_size = 2 bytes
                | (1u << 20)     // pad_enable
                | (3u << 22)     // pad_interval: 16 DWORDs (one 64B tile row)
                | (3u << 25));   // pad_amount: 4 DWORDs (16B)
    g1[1] = (int)((ld & 0xffffu) << 16);                               // tensor_dim0[15:0]
    g1[2] = (int)(((ld >> 16) & 0xffffu) | ((tile_rows & 0xffffu) << 16)); // dim0 hi | tensor_dim1 lo
    g1[3] = (int)(((tile_rows >> 16) & 0xffffu) | ((tile_k & 0xffffu) << 16)); // dim1 hi | tile_dim0
    g1[4] = (int)(tile_rows & 0xffffu);                                // tile_dim1 (tile_dim2=0)
    g1[5] = (int)(ld);                                                 // tensor_dim0_stride[31:0]
    g1[6] = 0;                                                         // stride hi | dim1_stride lo
    g1[7] = 0;
    const v4i z4 = {0, 0, 0, 0};
    const v8i z8 = {0, 0, 0, 0, 0, 0, 0, 0};
    __builtin_amdgcn_tensor_load_to_lds(g0, g1, z4, z4, z8, 0);
}
#endif

// ---------------------------------------------------------------------------
// WMMA GEMM: C[128 x 64-per-nblock] = A[M x K](bf16) * Wt_block[64 x K](bf16,[n][k]) + bias
// Double-buffered LDS; tiles staged by the Tensor Data Mover (wave 0 issues,
// s_wait_tensorcnt + workgroup barrier publishes). 8 waves, 16x64 strip each.
// ---------------------------------------------------------------------------
template<bool OUT_IS_F32>
__global__ __launch_bounds__(256)
void gemm128x64(const bf16t* __restrict__ A, int lda,
                const bf16t* __restrict__ Wt, long wBlockStride, int ldw,
                const float* __restrict__ bias,
                float* __restrict__ C32, bf16t* __restrict__ C16, int ldc,
                int K)
{
    __shared__ bf16t lA[2][128][40];   // 80B pitch (TDM pad / manual pad)
    __shared__ bf16t lB[2][64][40];

    const int tid  = threadIdx.x;
    const int m0   = blockIdx.x * 128;
    const int nb   = blockIdx.y;
    const bf16t* Ab = A + (long)m0 * lda;
    const bf16t* Bb = Wt + (long)nb * wBlockStride;
    const float* biasb = bias + nb * 64;
    const int col0 = nb * 64;

    const int wid  = tid >> 5;
    const int lane = tid & 31;
    const int lrow = lane & 15;
    const int hi   = (lane >> 4) & 1;

    v8f acc[4] = {};
    const int niter = K >> 5;

#if HAVE_TDM
    const unsigned ldsA0 = (unsigned)(size_t)&lA[0][0][0];
    const unsigned ldsA1 = (unsigned)(size_t)&lA[1][0][0];
    const unsigned ldsB0 = (unsigned)(size_t)&lB[0][0][0];
    const unsigned ldsB1 = (unsigned)(size_t)&lB[1][0][0];
    if (wid == 0) {   // prologue: stage tile 0 into buffer 0
        tdm_load_2d(ldsA0, Ab, 32u, 128u, (unsigned)lda);
        tdm_load_2d(ldsB0, Bb, 32u, 64u,  (unsigned)ldw);
    }
#endif

    for (int it = 0; it < niter; ++it) {
        const int buf = it & 1;
#if HAVE_TDM
        if (wid == 0) {
            if (it + 1 < niter) {   // stage next tile into the other buffer
                const int k0n = (it + 1) << 5;
                tdm_load_2d((it & 1) ? ldsA0 : ldsA1, Ab + k0n, 32u, 128u, (unsigned)lda);
                tdm_load_2d((it & 1) ? ldsB0 : ldsB1, Bb + k0n, 32u, 64u,  (unsigned)ldw);
                __builtin_amdgcn_s_wait_tensorcnt(2);   // current tile's pair done
            } else {
                __builtin_amdgcn_s_wait_tensorcnt(0);
            }
        }
        __syncthreads();            // publish current tile to all waves
#else
        const int k0 = it << 5;
        {   // manual bf16 tile copy fallback (pure moves, no converts)
            #pragma unroll
            for (int j = 0; j < 2; ++j) {
                const int idx = tid + j * 256;       // 512 8-elem chunks
                const int row = idx >> 2;
                const int c8  = (idx & 3) << 3;
                *(v8bf*)&lA[buf][row][c8] = *(const v8bf*)(Ab + (long)row * lda + k0 + c8);
            }
            const int row = tid >> 2;                // 256 chunks for B
            const int c8  = (tid & 3) << 3;
            *(v8bf*)&lB[buf][row][c8] = *(const v8bf*)(Bb + (long)row * ldw + k0 + c8);
        }
        __syncthreads();
#endif

        // ---- A fragment (16x32): lane<16 holds K{0-7,16-23}, lane>=16 K{8-15,24-31}
        const int arow = wid * 16 + lrow;
        const int ka   = hi ? 8 : 0;
        v8bf alo = *(const v8bf*)&lA[buf][arow][ka];
        v8bf ahi = *(const v8bf*)&lA[buf][arow][ka + 16];
        v16bf afrag;
        #pragma unroll
        for (int i = 0; i < 8; ++i) { afrag[i] = alo[i]; afrag[i + 8] = ahi[i]; }

        // ---- 4 WMMAs over the wave's 16x64 strip
        #pragma unroll
        for (int c = 0; c < 4; ++c) {
            const int bcol = c * 16 + lrow;
            const int kb   = hi ? 8 : 0;
            v8bf blo = *(const v8bf*)&lB[buf][bcol][kb];
            v8bf bhi = *(const v8bf*)&lB[buf][bcol][kb + 16];
            v16bf bfrag;
            #pragma unroll
            for (int i = 0; i < 8; ++i) { bfrag[i] = blo[i]; bfrag[i + 8] = bhi[i]; }
            acc[c] = __builtin_amdgcn_wmma_f32_16x16x32_bf16(
                false, afrag, false, bfrag, (short)0, acc[c], false, false);
        }
        __syncthreads();            // all readers done before buffer reuse
    }

    // ---- epilogue: C layout VGPR r -> rows r (lanes 0-15) / r+8 (lanes 16-31)
    const int mb = wid * 16 + (hi ? 8 : 0);
    #pragma unroll
    for (int c = 0; c < 4; ++c) {
        const int   col = col0 + c * 16 + lrow;
        const float bb  = biasb[c * 16 + lrow];
        #pragma unroll
        for (int r = 0; r < 8; ++r) {
            const long row = m0 + mb + r;
            const float val = acc[c][r] + bb;
            if (OUT_IS_F32) C32[row * ldc + col] = val;
            else            C16[row * ldc + col] = (bf16t)val;
        }
    }
}

// ---------------------------------------------------------------------------
// Banded attention, one thread per (b,h,s) query; streaming online softmax.
// 33 keys: 31 window keys (OOB rows collapse to bk/bv with multiplicity),
// + pos key + fsq key. scale = 1/sqrt(64).
// ---------------------------------------------------------------------------
__global__ __launch_bounds__(256)
void attn_win(const bf16t* __restrict__ Q,
              const bf16t* __restrict__ Kx, const bf16t* __restrict__ Vx,
              const bf16t* __restrict__ Kp, const bf16t* __restrict__ Vp,
              const bf16t* __restrict__ Kf, const bf16t* __restrict__ Vf,
              const float* __restrict__ bk, const float* __restrict__ bv,
              bf16t* __restrict__ O)
{
    const int gid = blockIdx.x * blockDim.x + threadIdx.x;
    if (gid >= B_ * H_ * S_) return;
    const int s = gid & (S_ - 1);
    const int h = (gid >> 11) & (H_ - 1);
    const int b = gid >> 15;

    const bf16t* qrow = Q + (long)s * 1024 + h * 64;
    float q[64];
    #pragma unroll
    for (int e = 0; e < 64; ++e) q[e] = (float)qrow[e];

    const float scale = 0.125f;   // 1/sqrt(HD)
    float m = -3.0e38f, l = 0.0f;
    float acc[64];
    #pragma unroll
    for (int e = 0; e < 64; ++e) acc[e] = 0.0f;

    auto process = [&](const bf16t* krow, const bf16t* vrow) {
        float sd = 0.0f;
        #pragma unroll
        for (int e = 0; e < 64; ++e) sd = fmaf((float)krow[e], q[e], sd);
        sd *= scale;
        const float mn   = fmaxf(m, sd);
        const float corr = __expf(m - mn);
        const float p    = __expf(sd - mn);
        l = l * corr + p;
        #pragma unroll
        for (int e = 0; e < 64; ++e) acc[e] = fmaf(p, (float)vrow[e], acc[e] * corr);
        m = mn;
    };

    // out-of-range window rows: kv row is zeros -> key = bk[h], value = bv[h]
    const int nl = (s < 15) ? (15 - s) : 0;
    const int nr = (s > S_ - 16) ? (s - (S_ - 16)) : 0;
    const int noob = nl + nr;
    if (noob > 0) {
        const float* kb = bk + h * 64;
        const float* vb = bv + h * 64;
        float sd = 0.0f;
        #pragma unroll
        for (int e = 0; e < 64; ++e) sd = fmaf(kb[e], q[e], sd);
        sd *= scale;
        const float mn   = fmaxf(m, sd);
        const float corr = __expf(m - mn);
        const float p    = (float)noob * __expf(sd - mn);
        l = l * corr + p;
        #pragma unroll
        for (int e = 0; e < 64; ++e) acc[e] = fmaf(p, vb[e], acc[e] * corr);
        m = mn;
    }

    const int t0 = (s - 15 < 0) ? 0 : s - 15;
    const int t1 = (s + 15 >= S_) ? S_ - 1 : s + 15;
    for (int t = t0; t <= t1; ++t) {
        const long r = (long)(b * S_ + t) * 1024 + h * 64;
        process(Kx + r, Vx + r);
    }
    {
        const long r = (long)s * 1024 + h * 64;
        process(Kp + r, Vp + r);
        process(Kf + r, Vf + r);
    }

    const float inv = 1.0f / l;
    bf16t* orow = O + (long)(b * S_ + s) * 1024 + h * 64;
    #pragma unroll
    for (int e = 0; e < 64; ++e) orow[e] = (bf16t)(acc[e] * inv);
}

// ---------------------------------------------------------------------------
extern "C" void kernel_launch(void* const* d_in, const int* in_sizes, int n_in,
                              void* d_out, int out_size, void* d_ws, size_t ws_size,
                              hipStream_t stream)
{
    (void)in_sizes; (void)n_in; (void)out_size; (void)ws_size;
    const float* x   = (const float*)d_in[0];
    const float* fsq = (const float*)d_in[1];
    const float* pos = (const float*)d_in[2];
    const float* Wq  = (const float*)d_in[3];
    const float* bq  = (const float*)d_in[4];
    const float* Wk  = (const float*)d_in[5];
    const float* bk  = (const float*)d_in[6];
    const float* Wv  = (const float*)d_in[7];
    const float* bv  = (const float*)d_in[8];
    const float* Wo  = (const float*)d_in[9];
    const float* bo  = (const float*)d_in[10];
    float* out = (float*)d_out;

    // ws layout (bf16): xb 8M | fb 4M | pb 4M | Wq 2M | Wk 2M | Wv 2M | Wo 2M |
    //                   Q 4M | Kx 8M | Vx 8M | Kp 4 | Vp 4 | Kf 4 | Vf 4 | Cc 8  (= 68 MB)
    char* ws = (char*)d_ws;
    bf16t* xb  = (bf16t*)(ws);
    bf16t* fb  = (bf16t*)(ws + ( 8ul << 20));
    bf16t* pb  = (bf16t*)(ws + (12ul << 20));
    bf16t* Wqb = (bf16t*)(ws + (16ul << 20));
    bf16t* Wkb = (bf16t*)(ws + (18ul << 20));
    bf16t* Wvb = (bf16t*)(ws + (20ul << 20));
    bf16t* Wob = (bf16t*)(ws + (22ul << 20));
    bf16t* Qw  = (bf16t*)(ws + (24ul << 20));
    bf16t* Kx  = (bf16t*)(ws + (28ul << 20));
    bf16t* Vx  = (bf16t*)(ws + (36ul << 20));
    bf16t* Kp  = (bf16t*)(ws + (44ul << 20));
    bf16t* Vp  = (bf16t*)(ws + (48ul << 20));
    bf16t* Kf  = (bf16t*)(ws + (52ul << 20));
    bf16t* Vf  = (bf16t*)(ws + (56ul << 20));
    bf16t* Cc  = (bf16t*)(ws + (60ul << 20));

    const dim3 blk(256);

    // -- pre-convert activations to bf16 (pure streaming) --
    cvt_f32_bf16<<<dim3((B_*S_*D_)/2048), blk, 0, stream>>>(x,   xb, B_*S_*D_);
    cvt_f32_bf16<<<dim3((S_*D_)/2048),    blk, 0, stream>>>(fsq, fb, S_*D_);
    cvt_f32_bf16<<<dim3((S_*D_)/2048),    blk, 0, stream>>>(pos, pb, S_*D_);
    // -- pre-convert + transpose weights to bf16 [n][k] per head / n-block --
    cvt_transpose<<<dim3((D_*HD_)/256, H_), blk, 0, stream>>>(Wq, Wqb, D_, HD_);
    cvt_transpose<<<dim3((D_*HD_)/256, H_), blk, 0, stream>>>(Wk, Wkb, D_, HD_);
    cvt_transpose<<<dim3((D_*HD_)/256, H_), blk, 0, stream>>>(Wv, Wvb, D_, HD_);
    cvt_transpose<<<dim3((D_*E_)/256,  1 ), blk, 0, stream>>>(Wo, Wob, D_, E_);

    const long WSTRIDE = 64L * 1024L;   // 64 transposed rows per n-block

    // -- Q/K/V projections (grid.y = head) --
    gemm128x64<false><<<dim3(S_/128,      H_), blk, 0, stream>>>(fb, D_, Wqb, WSTRIDE, D_, bq, nullptr, Qw, 1024, D_);
    gemm128x64<false><<<dim3((B_*S_)/128, H_), blk, 0, stream>>>(xb, D_, Wkb, WSTRIDE, D_, bk, nullptr, Kx, 1024, D_);
    gemm128x64<false><<<dim3((B_*S_)/128, H_), blk, 0, stream>>>(xb, D_, Wvb, WSTRIDE, D_, bv, nullptr, Vx, 1024, D_);
    gemm128x64<false><<<dim3(S_/128,      H_), blk, 0, stream>>>(pb, D_, Wkb, WSTRIDE, D_, bk, nullptr, Kp, 1024, D_);
    gemm128x64<false><<<dim3(S_/128,      H_), blk, 0, stream>>>(pb, D_, Wvb, WSTRIDE, D_, bv, nullptr, Vp, 1024, D_);
    gemm128x64<false><<<dim3(S_/128,      H_), blk, 0, stream>>>(fb, D_, Wkb, WSTRIDE, D_, bk, nullptr, Kf, 1024, D_);
    gemm128x64<false><<<dim3(S_/128,      H_), blk, 0, stream>>>(fb, D_, Wvb, WSTRIDE, D_, bv, nullptr, Vf, 1024, D_);

    // -- banded softmax attention -> concat (bf16) --
    attn_win<<<dim3((B_*H_*S_)/256), blk, 0, stream>>>(Qw, Kx, Vx, Kp, Vp, Kf, Vf, bk, bv, Cc);

    // -- output projection (4096x1024 @ Wo + bo) -> f32 out; grid.y = n-block --
    gemm128x64<true><<<dim3((B_*S_)/128, E_/64), blk, 0, stream>>>(Cc, 1024, Wob, WSTRIDE, D_, bo, out, nullptr, 1024, D_);
}
